// InLineAttention_44770739093824
// MI455X (gfx1250) — matrix operand order repaired and checked by reading, not compiled, not run on hardware
//
#include <hip/hip_runtime.h>
#include <hip/hip_bf16.h>
#include <cstdint>

typedef _Float16 half_t;
typedef __attribute__((ext_vector_type(16))) _Float16 v16h;
typedef __attribute__((ext_vector_type(8)))  float    v8f;
typedef __attribute__((ext_vector_type(4)))  unsigned int u32x4;
typedef __attribute__((ext_vector_type(8)))  int      i32x8;
typedef __attribute__((ext_vector_type(4)))  int      i32x4;

constexpr int kBT = 384;   // B*T
constexpr int kN  = 716;   // nodes
constexpr int kC  = 256;   // channels
constexpr int kH  = 8;     // heads
constexpr int kD  = 32;    // head dim
constexpr int kNT = 45;    // ceil(N/16) row tiles
constexpr float kScale = 0.17677669529663687f; // D^-0.5

__device__ __forceinline__ v8f wmma_f16(v16h a, v16h b, v8f c) {
  return __builtin_amdgcn_wmma_f32_16x16x32_f16(
      /*neg_a=*/false, a, /*neg_b=*/false, b,
      /*c_mod=*/(short)0, c, /*reuse_a=*/false, /*reuse_b=*/false);
}

// ---------------- TDM helpers (Tensor Data Mover) ----------------
__device__ __forceinline__ void tdm_wait0() {
#if __has_builtin(__builtin_amdgcn_s_wait_tensorcnt)
  __builtin_amdgcn_s_wait_tensorcnt((short)0);
#else
  asm volatile("s_wait_tensorcnt 0x0" ::: "memory");
#endif
}
__device__ __forceinline__ void tdm_wait2() {
#if __has_builtin(__builtin_amdgcn_s_wait_tensorcnt)
  __builtin_amdgcn_s_wait_tensorcnt((short)2);
#else
  asm volatile("s_wait_tensorcnt 0x2" ::: "memory");
#endif
}

// 2D f16 tile load, global -> LDS, packed LDS rows of tile_w elements.
// Rows >= tensor_h (and cols >= tensor_w) are zero-filled (TDM OOB semantics).
// Must be called by a single wave.
__device__ __forceinline__ void tdm_load_tile_f16(
    half_t* lds_dst, const half_t* gsrc,
    unsigned tensor_w, unsigned tensor_h,
    unsigned tile_w, unsigned tile_h,
    unsigned row_stride_elems) {
#if __has_builtin(__builtin_amdgcn_tensor_load_to_lds)
  unsigned lds_off = (unsigned)(uintptr_t)lds_dst;          // LDS byte offset (addr[31:0])
  unsigned long long ga = (unsigned long long)(uintptr_t)gsrc;
  // D# group 0: count=1 | lds_addr | global_addr[56:0] | type=2
  u32x4 g0 = { 1u, lds_off, (unsigned)ga,
               (unsigned)((ga >> 32) & 0x1FFFFFFu) | 0x80000000u };
  // D# group 1: data_size=1 (2B); tensor_dim0/1; tile_dim0/1; dim0 stride
  i32x8 g1 = { (int)0x10000u,
               (int)((tensor_w & 0xFFFFu) << 16),
               (int)((tensor_w >> 16) | ((tensor_h & 0xFFFFu) << 16)),
               (int)((tensor_h >> 16) | ((tile_w & 0xFFFFu) << 16)),
               (int)(tile_h & 0xFFFFu),
               (int)row_stride_elems,
               0, 0 };
  i32x4 z4 = { 0, 0, 0, 0 };
#if __clang_major__ >= 23
  i32x8 z8 = { 0, 0, 0, 0, 0, 0, 0, 0 };
  __builtin_amdgcn_tensor_load_to_lds(g0, g1, z4, z4, z8, 0);
#else
  __builtin_amdgcn_tensor_load_to_lds(g0, g1, z4, z4, 0);
#endif
#else
  // Fallback: wave-cooperative copy with zero fill.
  int lane = threadIdx.x & 31;
  for (unsigned r = 0; r < tile_h; ++r)
    for (unsigned c = lane; c < tile_w; c += 32)
      lds_dst[(size_t)r * tile_w + c] =
          (r < tensor_h && c < tensor_w) ? gsrc[(size_t)r * row_stride_elems + c]
                                         : (half_t)0.f;
#endif
}

// ---------------- WMMA fragment loaders ----------------
// A fragment: 16(M) x 32(K), row-major source, row = lane&15.
__device__ __forceinline__ v16h load_a16(const half_t* base, int ld) {
  int lane = threadIdx.x & 31;
  int row  = lane & 15;
  int kb   = (lane & 16) ? 8 : 0;
  const half_t* p = base + (size_t)row * ld;
  union { v16h v; half_t h[16]; } a;
#pragma unroll
  for (int j = 0; j < 4; ++j) {
    a.h[2*j]     = p[kb + 2*j];
    a.h[2*j + 1] = p[kb + 2*j + 1];
    a.h[8 + 2*j]     = p[16 + kb + 2*j];
    a.h[8 + 2*j + 1] = p[16 + kb + 2*j + 1];
  }
  return a.v;
}

// A fragment with transpose: A[row][k] = t[k*ld + row]  (for kv = k^T v)
__device__ __forceinline__ v16h load_aT16(const half_t* t, int ld) {
  int lane = threadIdx.x & 31;
  int row  = lane & 15;
  int kb   = (lane & 16) ? 8 : 0;
  union { v16h v; half_t h[16]; } a;
#pragma unroll
  for (int j = 0; j < 8; ++j) {
    int kk = (j < 4) ? (kb + 2*j) : (16 + kb + 2*(j - 4));
    a.h[2*j]     = t[(size_t)kk * ld + row];
    a.h[2*j + 1] = t[(size_t)(kk + 1) * ld + row];
  }
  return a.v;
}

// B fragment: 32(K) x 16(N), K-major source; lanes 0-15 K=0..7,16..23, 16-31 K=8..15,24..31.
__device__ __forceinline__ v16h load_b16(const half_t* t, int ld) {
  int lane = threadIdx.x & 31;
  int col  = lane & 15;
  int kofs = (lane & 16) ? 8 : 0;
  union { v16h v; half_t h[16]; } b;
#pragma unroll
  for (int i = 0; i < 16; ++i) b.h[i] = t[(size_t)(i < 8 ? kofs + i : 8 + kofs + i) * ld + col];
  return b.v;
}

// B fragment for W^T GEMMs: B[k][n] = W[nbase+n][kbase+k]; two 8-half runs (k=8h.., 16+8h..).
__device__ __forceinline__ v16h load_bT16(const half_t* wbase, int ld) {
  int lane = threadIdx.x & 31;
  const half_t* p = wbase + (size_t)(lane & 15) * ld + ((lane & 16) ? 8 : 0);
  union { v16h v; half_t h[16]; } b;
#pragma unroll
  for (int i = 0; i < 16; ++i) b.h[i] = p[i < 8 ? i : i + 8];
  return b.v;
}

// ---------------- kernel 0: weight f32 -> f16 ----------------
__global__ void cvt_kernel(const float* __restrict__ src, half_t* __restrict__ dst, int n) {
  int i = blockIdx.x * blockDim.x + threadIdx.x;
  if (i < n) dst[i] = (half_t)src[i];
}

// ---------------- kernel 1: q,k,v projections ----------------
__global__ void __launch_bounds__(512) qkv_kernel(
    const float* __restrict__ x, const half_t* __restrict__ W16,
    const float* __restrict__ bq, const float* __restrict__ bk, const float* __restrict__ bv,
    half_t* __restrict__ qbuf, half_t* __restrict__ kbuf, half_t* __restrict__ vbuf) {
  __shared__ half_t sx[16 * kC];
  int bt = blockIdx.x / kNT, nt = blockIdx.x % kNT;
  int tid = threadIdx.x;
  const float* xrow = x + (size_t)bt * kN * kC;
  for (int i = tid; i < 16 * kC; i += 512) {
    int rr = i >> 8, cc = i & 255;
    int n = nt * 16 + rr;
    int nc = n < kN ? n : kN - 1;
    sx[i] = (half_t)xrow[(size_t)nc * kC + cc];
  }
  if (nt + 1 < kNT && tid < 16) {
    int n = (nt + 1) * 16 + tid;
    if (n < kN) __builtin_prefetch(xrow + (size_t)n * kC, 0, 1);
  }
  __syncthreads();

  int w = tid >> 5, lane = tid & 31;
  int col = lane & 15, mofs = (lane & 16) ? 8 : 0;
  const half_t* Wmats[3] = { W16, W16 + 65536, W16 + 2 * 65536 };
  const float*  bias[3]  = { bq, bk, bv };
  half_t* outs[3]        = { qbuf, kbuf, vbuf };

  for (int mtx = 0; mtx < 3; ++mtx) {
    union { v8f v; float f[8]; } acc; acc.v = (v8f){};
#pragma unroll
    for (int ks = 0; ks < 8; ++ks) {
      v16h a = load_a16(sx + ks * 32, kC);
      v16h b = load_bT16(Wmats[mtx] + (size_t)(w * 16) * kC + ks * 32, kC);
      acc.v = wmma_f16(a, b, acc.v);
    }
    float bb = bias[mtx][w * 16 + col];
    half_t* dst = outs[mtx] + (size_t)bt * kN * kC;
#pragma unroll
    for (int r = 0; r < 8; ++r) {
      int n = nt * 16 + r + mofs;
      if (n < kN) dst[(size_t)n * kC + w * 16 + col] = (half_t)(acc.f[r] + bb);
    }
  }
}

// ---------------- kernel 2: kv = k^T v (TDM double-buffered), means ----------------
__global__ void __launch_bounds__(128) kv_kernel(
    const float* __restrict__ x,
    const half_t* __restrict__ kbuf, const half_t* __restrict__ vbuf,
    half_t* __restrict__ kvh, float* __restrict__ kmean, float* __restrict__ vmean,
    float* __restrict__ mfeat) {
  __shared__ half_t skbuf[2][32 * 32];
  __shared__ half_t svbuf[2][32 * 32];
  __shared__ float  sm[64];
  int bt = blockIdx.x >> 3, h = blockIdx.x & 7;
  int tid = threadIdx.x, lane = tid & 31, w = tid >> 5;
  const half_t* kp = kbuf + (size_t)bt * kN * kC + h * kD;
  const half_t* vp = vbuf + (size_t)bt * kN * kC + h * kD;
  constexpr int NTILE = (kN + 31) / 32;  // 23

  union { v8f v; float f[8]; } acc; acc.v = (v8f){};
  float ksum = 0.f, vsum = 0.f;
  int dq = w >> 1, eq = w & 1;

  if (w == 0) {  // prologue: tile 0 in flight
    tdm_load_tile_f16(skbuf[0], kp, kD, kN, kD, 32, kC);
    tdm_load_tile_f16(svbuf[0], vp, kD, kN, kD, 32, kC);
  }
  for (int it = 0; it < NTILE; ++it) {
    if (w == 0) {
      if (it + 1 < NTILE) {  // issue next tile, then wait for current (<=2 in flight)
        int n0 = (it + 1) * 32;
        tdm_load_tile_f16(skbuf[(it + 1) & 1], kp + (size_t)n0 * kC,
                          kD, (unsigned)(kN - n0), kD, 32, kC);
        tdm_load_tile_f16(svbuf[(it + 1) & 1], vp + (size_t)n0 * kC,
                          kD, (unsigned)(kN - n0), kD, 32, kC);
        tdm_wait2();
      } else {
        tdm_wait0();
      }
    }
    __syncthreads();
    const half_t* sk = skbuf[it & 1];
    const half_t* sv = svbuf[it & 1];
    v16h a = load_aT16(sk + dq * 16, 32);  // A[d][n] = k[n][d]
    v16h b = load_b16(sv + eq * 16, 32);   // B[n][e] = v[n][e]
    acc.v = wmma_f16(a, b, acc.v);
    if (tid < 32) {
      float s = 0.f;
      for (int rr = 0; rr < 32; ++rr) s += (float)sk[rr * 32 + tid];
      ksum += s;
    } else if (tid < 64) {
      int dd = tid - 32;
      float s = 0.f;
      for (int rr = 0; rr < 32; ++rr) s += (float)sv[rr * 32 + dd];
      vsum += s;
    }
    __syncthreads();
  }

  const float sc = kScale / (float)kN;
  int col = lane & 15, mofs = (lane & 16) ? 8 : 0;
  half_t* kvdst = kvh + ((size_t)bt * kH + h) * kD * kD;
#pragma unroll
  for (int r = 0; r < 8; ++r) {
    int d = dq * 16 + r + mofs, e = eq * 16 + col;
    kvdst[d * kD + e] = (half_t)(acc.f[r] * sc);
  }
  if (tid < 32) kmean[((size_t)bt * kH + h) * kD + tid] = ksum / (float)kN;
  else if (tid < 64) vmean[((size_t)bt * kH + h) * kD + (tid - 32)] = vsum / (float)kN;

  if (tid >= 64) {  // x mean over N for this head's channels (2 threads/channel)
    int dd = (tid - 64) & 31;
    int hs = (tid - 64) >> 5;
    const float* xp = x + (size_t)bt * kN * kC + h * kD + dd;
    float s = 0.f;
    for (int n = hs; n < kN; n += 2) s += xp[(size_t)n * kC];
    sm[tid - 64] = s;
  }
  __syncthreads();
  if (tid >= 64 && tid < 96) {
    int dd = tid - 64;
    mfeat[((size_t)bt * kH + h) * kD + dd] = (sm[dd] + sm[dd + 32]) / (float)kN;
  }
}

// ---------------- kernel 3: dynamic 3-tap weights (grouped MLP + GELU) ----------------
__global__ void __launch_bounds__(256) rw_kernel(
    const float* __restrict__ mfeat, const float* __restrict__ w1, const float* __restrict__ b1,
    const float* __restrict__ w2, const float* __restrict__ b2, float* __restrict__ rw) {
  __shared__ float smf[kC];
  __shared__ float sh1[kC];
  int bt = blockIdx.x, c = threadIdx.x;
  smf[c] = mfeat[(size_t)bt * kC + c];
  __syncthreads();
  int g = c >> 5, o = c & 31;
  float s = b1[c];
  const float* w1r = w1 + (size_t)c * kD;
  for (int i = 0; i < kD; ++i) s += smf[g * kD + i] * w1r[i];
  s = 0.5f * s * (1.f + erff(s * 0.70710678118654752f));  // exact GELU
  sh1[c] = s;
  __syncthreads();
  float* dst = rw + ((size_t)bt * kC + c) * 3;
#pragma unroll
  for (int t = 0; t < 3; ++t) {
    int row = g * 96 + 3 * o + t;  // h2[g][3*o+t] -> rw[c][t]
    float s2 = b2[row];
    const float* w2r = w2 + (size_t)row * kD;
    for (int i = 0; i < kD; ++i) s2 += sh1[g * kD + i] * w2r[i];
    dst[t] = s2;
  }
}

// ---------------- kernel 4: attn + correction + conv epilogue + output GEMM ----------------
__global__ void __launch_bounds__(512) attn_out_kernel(
    const half_t* __restrict__ qbuf, const half_t* __restrict__ vbuf,
    const half_t* __restrict__ kvh, const float* __restrict__ kmean,
    const float* __restrict__ vmean, const float* __restrict__ rw,
    const half_t* __restrict__ Wo16, const float* __restrict__ bo,
    float* __restrict__ out) {
  __shared__ half_t sq[16 * kC];       // q tile (TDM, zero-filled OOB rows)
  __shared__ half_t svc[18 * kC];      // v context rows n-1..n+16 (zero padded)
  __shared__ half_t skv[kH * kD * kD]; // per-head kv (TDM)
  __shared__ half_t sy[16 * kC];       // fused epilogue output (f16 for GEMM)
  __shared__ float  srw[kC * 3];
  __shared__ float  skm[kC];
  __shared__ float  svm[kC];
  __shared__ float  sqk[16 * kH];

  int bt = blockIdx.x / kNT, nt = blockIdx.x % kNT;
  int tid = threadIdx.x;
  const half_t* qp = qbuf + (size_t)bt * kN * kC;
  const half_t* vp = vbuf + (size_t)bt * kN * kC;

  if (tid < 32) {  // wave 0: async tensor loads of q tile + kv block
    tdm_load_tile_f16(sq, qp + (size_t)(nt * 16) * kC,
                      kC, (unsigned)(kN - nt * 16), kC, 16, kC);
    tdm_load_tile_f16(skv, kvh + (size_t)bt * kH * kD * kD,
                      kD * kD, kH, kD * kD, kH, kD * kD);
  }
  for (int i = tid; i < 18 * kC; i += 512) {  // overlapped manual staging
    int rr = i >> 8, cc = i & 255;
    int n = nt * 16 + rr - 1;
    svc[i] = (n >= 0 && n < kN) ? vp[(size_t)n * kC + cc] : (half_t)0.f;
  }
  for (int i = tid; i < kC * 3; i += 512) srw[i] = rw[(size_t)bt * kC * 3 + i];
  for (int i = tid; i < kC; i += 512) {
    skm[i] = kmean[(size_t)bt * kC + i];
    svm[i] = vmean[(size_t)bt * kC + i];
  }
  if (tid < 32) tdm_wait0();
  __syncthreads();

  if (tid < 128) {  // qk[n,h] = scale * dot(q[n, h*D:], kmean[h])
    int rr = tid >> 3, h = tid & 7;
    float s = 0.f;
    for (int d = 0; d < kD; ++d) s += (float)sq[rr * kC + h * kD + d] * skm[h * kD + d];
    sqk[rr * kH + h] = s * kScale;
  }
  __syncthreads();

  int w = tid >> 5, lane = tid & 31;
  int col = lane & 15, mofs = (lane & 16) ? 8 : 0;
  {
    int h = w >> 1, eq = w & 1;
    union { v8f v; float f[8]; } acc; acc.v = (v8f){};
    v16h a = load_a16(sq + h * kD, kC);                 // q[:, h*D + k]
    v16h b = load_b16(skv + h * kD * kD + eq * 16, kD); // kv[k][e]
    acc.v = wmma_f16(a, b, acc.v);
#pragma unroll
    for (int r = 0; r < 8; ++r) {
      int rr = r + mofs;
      int e = eq * 16 + col;
      int c = h * kD + e;
      float qk = sqk[rr * kH + h];
      float xa = acc.f[r] + (1.f - qk) * svm[c];
      float res = srw[c * 3 + 0] * (float)svc[(size_t)rr * kC + c]
                + srw[c * 3 + 1] * (float)svc[(size_t)(rr + 1) * kC + c]
                + srw[c * 3 + 2] * (float)svc[(size_t)(rr + 2) * kC + c];
      sy[rr * kC + c] = (half_t)(xa + res);
    }
  }
  __syncthreads();

  {  // out = y @ Wo^T + bo
    union { v8f v; float f[8]; } acc; acc.v = (v8f){};
#pragma unroll
    for (int ks = 0; ks < 8; ++ks) {
      v16h a = load_a16(sy + ks * 32, kC);
      v16h b = load_bT16(Wo16 + (size_t)(w * 16) * kC + ks * 32, kC);
      acc.v = wmma_f16(a, b, acc.v);
    }
    float bb = bo[w * 16 + col];
    float* dst = out + (size_t)bt * kN * kC;
#pragma unroll
    for (int r = 0; r < 8; ++r) {
      int n = nt * 16 + r + mofs;
      if (n < kN) dst[(size_t)n * kC + w * 16 + col] = acc.f[r] + bb;
    }
  }
}

extern "C" void kernel_launch(void* const* d_in, const int* in_sizes, int n_in,
                              void* d_out, int out_size, void* d_ws, size_t ws_size,
                              hipStream_t stream) {
  (void)in_sizes; (void)n_in; (void)out_size; (void)ws_size;
  const float* x  = (const float*)d_in[0];
  const float* Wq = (const float*)d_in[1];  const float* bq = (const float*)d_in[2];
  const float* Wk = (const float*)d_in[3];  const float* bk = (const float*)d_in[4];
  const float* Wv = (const float*)d_in[5];  const float* bv = (const float*)d_in[6];
  const float* Wo = (const float*)d_in[7];  const float* bo = (const float*)d_in[8];
  const float* w1 = (const float*)d_in[9];  const float* b1 = (const float*)d_in[10];
  const float* w2 = (const float*)d_in[11]; const float* b2 = (const float*)d_in[12];
  float* out = (float*)d_out;

  const size_t SZ = (size_t)kBT * kN * kC;  // 70,385,664 elements
  half_t* W16  = (half_t*)d_ws;                 // 4 * 65536 halfs
  half_t* qbuf = W16 + 4 * 65536;               // SZ halfs
  half_t* kbuf = qbuf + SZ;                     // SZ halfs
  half_t* vbuf = kbuf + SZ;                     // SZ halfs
  half_t* kvh  = vbuf + SZ;                     // kBT*kH*kD*kD halfs
  float*  fbase = (float*)(kvh + (size_t)kBT * kH * kD * kD);
  float*  kmean = fbase;                        // kBT*kC
  float*  vmean = kmean + (size_t)kBT * kC;     // kBT*kC
  float*  mfeat = vmean + (size_t)kBT * kC;     // kBT*kC
  float*  rwbuf = mfeat + (size_t)kBT * kC;     // kBT*kC*3

  cvt_kernel<<<256, 256, 0, stream>>>(Wq, W16 + 0 * 65536, 65536);
  cvt_kernel<<<256, 256, 0, stream>>>(Wk, W16 + 1 * 65536, 65536);
  cvt_kernel<<<256, 256, 0, stream>>>(Wv, W16 + 2 * 65536, 65536);
  cvt_kernel<<<256, 256, 0, stream>>>(Wo, W16 + 3 * 65536, 65536);

  qkv_kernel<<<kBT * kNT, 512, 0, stream>>>(x, W16, bq, bk, bv, qbuf, kbuf, vbuf);
  kv_kernel<<<kBT * kH, 128, 0, stream>>>(x, kbuf, vbuf, kvh, kmean, vmean, mfeat);
  rw_kernel<<<kBT, 256, 0, stream>>>(mfeat, w1, b1, w2, b2, rwbuf);
  attn_out_kernel<<<kBT * kNT, 512, 0, stream>>>(qbuf, vbuf, kvh, kmean, vmean, rwbuf,
                                                 W16 + 3 * 65536, bo, out);
}
